// PatchRNN_56959856280339
// MI455X (gfx1250) — matrix operand order repaired and checked by reading, not compiled, not run on hardware
//
#include <hip/hip_runtime.h>
#include <hip/hip_bf16.h>
#include <math.h>

// ---------------------------------------------------------------------------
// PatchRNN (4-layer GRU) for MI455X / gfx1250, bf16 WMMA + fp32 gates.
// ---------------------------------------------------------------------------

typedef __attribute__((ext_vector_type(16))) __bf16 v16bf;
typedef __attribute__((ext_vector_type(8)))  __bf16 v8bf;
typedef __attribute__((ext_vector_type(8)))  float  v8f;

#define GB   32          // batch
#define GS   512         // seq len
#define GD   768         // model dim
#define GH   1024        // hidden
#define GL   4           // layers
#define G3H  3072        // 3*H

// ---- WMMA helpers ----------------------------------------------------------

__device__ __forceinline__ v8f wmma_bf16(v16bf a, v16bf b, v8f c) {
  return __builtin_amdgcn_wmma_f32_16x16x32_bf16(
      /*neg_a=*/false, a, /*neg_b=*/false, b,
      /*c_mod=*/(short)0, c, /*reuse_a=*/false, /*reuse_b=*/false);
}

// A fragment (16x32 bf16, MxK). ISA layout: lane-group g, half j ->
//   K = (j<8) ? 8g+j : 16+8g+(j-8).  Two contiguous b128 loads.
__device__ __forceinline__ v16bf load_a_frag(const __bf16* __restrict__ row,
                                             int k0, int g) {
  v8bf lo = *(const v8bf*)(row + k0 + 8 * g);
  v8bf hi = *(const v8bf*)(row + k0 + 16 + 8 * g);
  v16bf a;
#pragma unroll
  for (int i = 0; i < 8; ++i) { a[i] = lo[i]; a[8 + i] = hi[i]; }
  return a;
}

// B fragment (32x16 bf16, KxN), B[k,n] = W[n,k] with W row-major [N x ldk].
// lane-group g, half j -> K = (j<8) ? 8g+j : 16+8g+(j-8). 'wrow' already points at W row n.
__device__ __forceinline__ v16bf load_b_frag(const __bf16* __restrict__ wrow,
                                             int k0, int g) {
  v8bf lo = *(const v8bf*)(wrow + k0 + 8 * g);
  v8bf hi = *(const v8bf*)(wrow + k0 + 16 + 8 * g);
  v16bf b;
#pragma unroll
  for (int i = 0; i < 8; ++i) { b[i] = lo[i]; b[8 + i] = hi[i]; }
  return b;
}

__device__ __forceinline__ float sigmoidf_(float x) {
  return 1.0f / (1.0f + __expf(-x));
}

// ---- prep kernels ----------------------------------------------------------

__global__ __launch_bounds__(256) void k_cvt_bf16(const float* __restrict__ s,
                                                  __bf16* __restrict__ d, int n) {
  int i = blockIdx.x * 256 + threadIdx.x;
  if (i < n) d[i] = (__bf16)s[i];
}

// xp_bf[(t*32+b)*768 + d] = x[b,t,d] + pos[t,d]   (time-major for GEMM rows)
__global__ __launch_bounds__(256) void k_prep_xp(const float* __restrict__ x,
                                                 const float* __restrict__ pos,
                                                 __bf16* __restrict__ xp) {
  int i = blockIdx.x * 256 + threadIdx.x;
  if (i >= GS * GB * GD) return;
  int d = i % GD;
  int r = i / GD;
  int b = r & (GB - 1);
  int t = r >> 5;
  float v = x[((size_t)b * GS + t) * GD + d] + pos[(size_t)t * GD + d];
  xp[i] = (__bf16)v;
}

__global__ __launch_bounds__(256) void k_zero_h(float* __restrict__ hf,
                                                __bf16* __restrict__ hb) {
  int i = blockIdx.x * 256 + threadIdx.x;
  if (i < GL * 2 * GB * GH) { hf[i] = 0.0f; hb[i] = (__bf16)0.0f; }
}

// ---- layer-0 input projection: gi0 = xp @ Wih0^T + bih0  [16384 x 3072] ----

__global__ __launch_bounds__(256) void k_gemm_gi0(const __bf16* __restrict__ xp,
                                                  const __bf16* __restrict__ W0,
                                                  const float* __restrict__ bih0,
                                                  float* __restrict__ gi0) {
  int gw   = blockIdx.x * 8 + (threadIdx.x >> 5);   // global wave id
  int lane = threadIdx.x & 31;
  int g = lane >> 4, ln = lane & 15;
  int ntile = gw % (G3H / 16);                      // 192
  int mtile = gw / (G3H / 16);                      // 1024
  const __bf16* arow = xp + (size_t)(mtile * 16 + ln) * GD;
  const __bf16* brow = W0 + (size_t)(ntile * 16 + ln) * GD;
  v8f acc = {0, 0, 0, 0, 0, 0, 0, 0};
#pragma unroll 4
  for (int k0 = 0; k0 < GD; k0 += 32)
    acc = wmma_bf16(load_a_frag(arow, k0, g), load_b_frag(brow, k0, g), acc);
  int nc = ntile * 16 + ln;
  float bias = bih0[nc];
#pragma unroll
  for (int v = 0; v < 8; ++v) {
    int m = mtile * 16 + v + 8 * g;                 // C/D layout: M = v + 8*(lane/16)
    gi0[(size_t)m * G3H + nc] = acc[v] + bias;
  }
}

// ---- pipelined GRU step: launch i computes layer l at time t = i - l -------
// grid = 512 blocks: [l(4)][mtile(2)][ntile(64)], 128 threads (4 waves, K-split)

__global__ __launch_bounds__(128) void k_gru_pipe(
    int i,
    const __bf16* __restrict__ Wih_bf,   // [L-1][3H][H]
    const __bf16* __restrict__ Whh_bf,   // [L][3H][H]
    const float*  __restrict__ bih_all,  // [L][3H]
    const float*  __restrict__ bhh_all,  // [L][3H]
    const float*  __restrict__ gi0,      // [S*B][3H]
    float*        __restrict__ h_f32,    // [L][2][B][H]
    __bf16*       __restrict__ h_bf,     // [L][2][B][H]
    __bf16*       __restrict__ hist) {   // [S][B][H]  (layer-3 outputs)
  int l     = blockIdx.x >> 7;
  int rem   = blockIdx.x & 127;
  int mtile = rem >> 6;
  int ntile = rem & 63;
  int t = i - l;
  if (t < 0 || t >= GS) return;

  int tid = threadIdx.x;
  int w = tid >> 5, lane = tid & 31;
  int g = lane >> 4, ln = lane & 15;
  int p = t & 1, pp = p ^ 1;

  const __bf16* Whh_l = Whh_bf + (size_t)l * G3H * GH;
  const __bf16* Wih_l = Wih_bf + (size_t)(l > 0 ? l - 1 : 0) * G3H * GH;
  const __bf16* hprev_bf = h_bf  + ((size_t)l * 2 + pp) * GB * GH;
  const float*  hprev_f  = h_f32 + ((size_t)l * 2 + pp) * GB * GH;
  const __bf16* inp_bf   = h_bf  + ((size_t)(l > 0 ? l - 1 : 0) * 2 + p) * GB * GH;
  float*  hout_f  = h_f32 + ((size_t)l * 2 + p) * GB * GH;
  __bf16* hout_bf = h_bf  + ((size_t)l * 2 + p) * GB * GH;
  const float* bih = bih_all + l * G3H;
  const float* bhh = bhh_all + l * G3H;
  const float* gi0_t = gi0 + (size_t)t * GB * G3H;

  int arow = mtile * 16 + ln;
  const __bf16* ah_base = hprev_bf + (size_t)arow * GH;
  const __bf16* ai_base = inp_bf   + (size_t)arow * GH;
  int wn = ntile * 16 + ln;                       // weight row (gate column)
  const __bf16* whr = Whh_l + (size_t)(wn)            * GH;
  const __bf16* whz = Whh_l + (size_t)(GH + wn)       * GH;
  const __bf16* whn = Whh_l + (size_t)(2 * GH + wn)   * GH;
  const __bf16* wir = Wih_l + (size_t)(wn)            * GH;
  const __bf16* wiz = Wih_l + (size_t)(GH + wn)       * GH;
  const __bf16* win = Wih_l + (size_t)(2 * GH + wn)   * GH;

  v8f accR  = {0, 0, 0, 0, 0, 0, 0, 0};
  v8f accZ  = {0, 0, 0, 0, 0, 0, 0, 0};
  v8f accNi = {0, 0, 0, 0, 0, 0, 0, 0};
  v8f accNh = {0, 0, 0, 0, 0, 0, 0, 0};

  int kstart = w * 256;                           // K-split across 4 waves
#pragma unroll 2
  for (int kb = 0; kb < 8; ++kb) {
    int k0 = kstart + kb * 32;
    v16bf ah = load_a_frag(ah_base, k0, g);
    accR  = wmma_bf16(ah, load_b_frag(whr, k0, g), accR);
    accZ  = wmma_bf16(ah, load_b_frag(whz, k0, g), accZ);
    accNh = wmma_bf16(ah, load_b_frag(whn, k0, g), accNh);
    if (l > 0) {
      v16bf ai = load_a_frag(ai_base, k0, g);
      accR  = wmma_bf16(ai, load_b_frag(wir, k0, g), accR);
      accZ  = wmma_bf16(ai, load_b_frag(wiz, k0, g), accZ);
      accNi = wmma_bf16(ai, load_b_frag(win, k0, g), accNi);
    }
  }

  // K-split reduction through LDS, then gate math on the 16x16 tile.
  __shared__ float red[4][4][256];
#pragma unroll
  for (int v = 0; v < 8; ++v) {
    red[w][0][v * 32 + lane] = accR[v];
    red[w][1][v * 32 + lane] = accZ[v];
    red[w][2][v * 32 + lane] = accNi[v];
    red[w][3][v * 32 + lane] = accNh[v];
  }
  __syncthreads();

  for (int e = tid; e < 256; e += 128) {
    int lv = e >> 5, le = e & 31;
    int M = lv + 8 * (le >> 4);
    int N = le & 15;
    int m  = mtile * 16 + M;
    int nc = ntile * 16 + N;
    float R = 0.f, Z = 0.f, Ni = 0.f, Nh = 0.f;
#pragma unroll
    for (int ww = 0; ww < 4; ++ww) {
      R  += red[ww][0][e];
      Z  += red[ww][1][e];
      Ni += red[ww][2][e];
      Nh += red[ww][3][e];
    }
    float in_v;
    if (l == 0) {                                  // gi0 already has bih
      const float* gr = gi0_t + (size_t)m * G3H;
      R   += gr[nc]          + bhh[nc];
      Z   += gr[GH + nc]     + bhh[GH + nc];
      in_v = gr[2 * GH + nc];
    } else {
      R   += bih[nc]          + bhh[nc];
      Z   += bih[GH + nc]     + bhh[GH + nc];
      in_v = Ni + bih[2 * GH + nc];
    }
    float hn = Nh + bhh[2 * GH + nc];
    float r = sigmoidf_(R);
    float z = sigmoidf_(Z);
    float nv = tanhf(in_v + r * hn);
    float hp = hprev_f[(size_t)m * GH + nc];
    float hnew = (1.0f - z) * nv + z * hp;
    hout_f[(size_t)m * GH + nc]  = hnew;
    hout_bf[(size_t)m * GH + nc] = (__bf16)hnew;
    if (l == 3) hist[((size_t)t * GB + m) * GH + nc] = (__bf16)hnew;
  }
}

// ---- output projection: out[b,t,:] = hist[t,b,:] @ Wout^T + bout -----------

__global__ __launch_bounds__(256) void k_gemm_out(const __bf16* __restrict__ hist,
                                                  const __bf16* __restrict__ Wout,
                                                  const float* __restrict__ bout,
                                                  float* __restrict__ out) {
  int gw   = blockIdx.x * 8 + (threadIdx.x >> 5);
  int lane = threadIdx.x & 31;
  int g = lane >> 4, ln = lane & 15;
  int ntile = gw % (GD / 16);                     // 48
  int mtile = gw / (GD / 16);                     // 1024
  const __bf16* arow = hist + (size_t)(mtile * 16 + ln) * GH;
  const __bf16* brow = Wout + (size_t)(ntile * 16 + ln) * GH;
  v8f acc = {0, 0, 0, 0, 0, 0, 0, 0};
#pragma unroll 4
  for (int k0 = 0; k0 < GH; k0 += 32)
    acc = wmma_bf16(load_a_frag(arow, k0, g), load_b_frag(brow, k0, g), acc);
  int nc = ntile * 16 + ln;
  float bias = bout[nc];
#pragma unroll
  for (int v = 0; v < 8; ++v) {
    int m = mtile * 16 + v + 8 * g;               // row over (t,b)
    int t = m >> 5, b = m & (GB - 1);
    out[((size_t)b * GS + t) * GD + nc] = acc[v] + bias;
  }
}

// ---- hs_final copy: d_out tail [L,B,H] <- h_f32[l][parity=1] ---------------

__global__ __launch_bounds__(256) void k_hsfinal(const float* __restrict__ h_f32,
                                                 float* __restrict__ out_tail) {
  int i = blockIdx.x * 256 + threadIdx.x;
  if (i >= GL * GB * GH) return;
  int l = i >> 15;              // 32*1024 = 32768 per layer
  int rem = i & 32767;
  out_tail[i] = h_f32[((size_t)l * 2 + 1) * GB * GH + rem];  // (S-1)&1 == 1
}

// ---------------------------------------------------------------------------

extern "C" void kernel_launch(void* const* d_in, const int* in_sizes, int n_in,
                              void* d_out, int out_size, void* d_ws, size_t ws_size,
                              hipStream_t stream) {
  (void)in_sizes; (void)n_in; (void)out_size; (void)ws_size;

  const float* x    = (const float*)d_in[0];   // [B,S,D]
  const float* pos  = (const float*)d_in[1];   // [1,MAXSEQ,D]
  const float* Wih0 = (const float*)d_in[2];   // [3H,D]
  const float* Wih  = (const float*)d_in[3];   // [L-1,3H,H]
  const float* Whh  = (const float*)d_in[4];   // [L,3H,H]
  const float* bih  = (const float*)d_in[5];   // [L,3H]
  const float* bhh  = (const float*)d_in[6];   // [L,3H]
  const float* Wout = (const float*)d_in[7];   // [D,H]
  const float* bout = (const float*)d_in[8];   // [D]
  float* out = (float*)d_out;

  // workspace carve-up (256B aligned)
  size_t off = 0;
  auto carve = [&](size_t bytes) -> char* {
    char* p = (char*)d_ws + off;
    off += (bytes + 255) & ~(size_t)255;
    return p;
  };
  const size_t nWih0 = (size_t)G3H * GD;
  const size_t nWih  = (size_t)(GL - 1) * G3H * GH;
  const size_t nWhh  = (size_t)GL * G3H * GH;
  const size_t nWout = (size_t)GD * GH;
  const size_t nXp   = (size_t)GS * GB * GD;
  const size_t nHist = (size_t)GS * GB * GH;
  const size_t nHbuf = (size_t)GL * 2 * GB * GH;
  const size_t nGi0  = (size_t)GS * GB * G3H;

  __bf16* Wih0_bf = (__bf16*)carve(nWih0 * 2);
  __bf16* Wih_bf  = (__bf16*)carve(nWih * 2);
  __bf16* Whh_bf  = (__bf16*)carve(nWhh * 2);
  __bf16* Wout_bf = (__bf16*)carve(nWout * 2);
  __bf16* xp_bf   = (__bf16*)carve(nXp * 2);
  __bf16* hist_bf = (__bf16*)carve(nHist * 2);
  __bf16* h_bf    = (__bf16*)carve(nHbuf * 2);
  float*  h_f32   = (float*)carve(nHbuf * 4);
  float*  gi0     = (float*)carve(nGi0 * 4);

  // 1) convert weights to bf16, build time-major (x+pos) in bf16, zero h
  k_cvt_bf16<<<(int)((nWih0 + 255) / 256), 256, 0, stream>>>(Wih0, Wih0_bf, (int)nWih0);
  k_cvt_bf16<<<(int)((nWih  + 255) / 256), 256, 0, stream>>>(Wih,  Wih_bf,  (int)nWih);
  k_cvt_bf16<<<(int)((nWhh  + 255) / 256), 256, 0, stream>>>(Whh,  Whh_bf,  (int)nWhh);
  k_cvt_bf16<<<(int)((nWout + 255) / 256), 256, 0, stream>>>(Wout, Wout_bf, (int)nWout);
  k_prep_xp<<<(int)((nXp + 255) / 256), 256, 0, stream>>>(x, pos, xp_bf);
  k_zero_h<<<(int)((nHbuf + 255) / 256), 256, 0, stream>>>(h_f32, h_bf);

  // 2) layer-0 input projection over all timesteps (parallel-in-time)
  //    16384 x 3072 GEMM, K=768 -> 1024*192 tiles / 8 waves per block
  k_gemm_gi0<<<(16384 / 16) * (G3H / 16) / 8, 256, 0, stream>>>(xp_bf, Wih0_bf, bih, gi0);

  // 3) wavefront-pipelined recurrence: launch i does layer l at t = i - l
  for (int i = 0; i < GS + GL - 1; ++i)
    k_gru_pipe<<<GL * 2 * (GH / 16), 128, 0, stream>>>(
        i, Wih_bf, Whh_bf, bih, bhh, gi0, h_f32, h_bf, hist_bf);

  // 4) output projection: 16384 x 768 GEMM, K=1024
  k_gemm_out<<<(16384 / 16) * (GD / 16) / 8, 256, 0, stream>>>(hist_bf, Wout_bf, bout, out);

  // 5) hs_final tail
  k_hsfinal<<<(GL * GB * GH) / 256, 256, 0, stream>>>(h_f32, out + (size_t)GB * GS * GD);
}
